// MoeMlp_49048526520563
// MI455X (gfx1250) — matrix operand order repaired and checked by reading, not compiled, not run on hardware
//
#include <hip/hip_runtime.h>

typedef __attribute__((ext_vector_type(16))) __bf16 v16bf;
typedef __attribute__((ext_vector_type(8)))  __bf16 v8bf;
typedef __attribute__((ext_vector_type(8)))  float  v8f;
typedef __attribute__((ext_vector_type(4)))  float  v4f;
typedef __attribute__((ext_vector_type(8)))  unsigned short v8us;

#define B_DIM 4096
#define DIN   1024
#define H_DIM 2048
#define C_DIM 1000
#define CPAD  1024
#define M_EXP 16
#define PITER 10
#define LPAD  40   // LDS row pitch in halfwords (80B: 16B-aligned, bank-conflict-free)

__device__ __forceinline__ unsigned short f2bf(float f) {
  unsigned u = __float_as_uint(f);
  u += 0x7FFFu + ((u >> 16) & 1u);           // round-to-nearest-even
  return (unsigned short)(u >> 16);
}

// CDNA5 async global->LDS copy (ASYNCcnt-tracked), per-lane 16B.
__device__ __forceinline__ void async_copy_b128(unsigned lds_addr, const void* gptr) {
  asm volatile("global_load_async_to_lds_b128 %0, %1, off"
               :: "v"(lds_addr), "v"((unsigned long long)(size_t)gptr)
               : "memory");
}
__device__ __forceinline__ void wait_async0() {
  asm volatile("s_wait_asynccnt 0x0" ::: "memory");
}

// ---------------- conversions ----------------
// plain f32 -> bf16, 8 elements/thread
__global__ void cvt_bf16x8_kernel(const float* __restrict__ in,
                                  unsigned short* __restrict__ out, long long n8) {
  long long i = (long long)blockIdx.x * blockDim.x + threadIdx.x;
  if (i >= n8) return;
  const v4f* p = (const v4f*)(in + i * 8);
  v4f a = p[0], b = p[1];
  v8us o;
#pragma unroll
  for (int j = 0; j < 4; ++j) { o[j] = f2bf(a[j]); o[j + 4] = f2bf(b[j]); }
  *(v8us*)(out + i * 8) = o;
}

// transpose + convert + pad: out[m][NP][K] (bf16) = in[m][K][N]^T, rows N..NP-1 = 0
__global__ __launch_bounds__(256)
void tr_cvt_kernel(const float* __restrict__ in, long long sIn,
                   unsigned short* __restrict__ out, long long sOut,
                   int K, int N, int NP) {
  __shared__ float tile[32][33];
  const int m  = blockIdx.z;
  const int n0 = blockIdx.x * 32;
  const int k0 = blockIdx.y * 32;
  const float* ip = in + (long long)m * sIn;
  unsigned short* op = out + (long long)m * sOut;
  const int tx = threadIdx.x & 31, ty = threadIdx.x >> 5;   // 32 x 8
#pragma unroll
  for (int q = 0; q < 4; ++q) {
    int k = k0 + ty + q * 8;
    int n = n0 + tx;
    tile[ty + q * 8][tx] = (n < N) ? ip[(long long)k * N + n] : 0.f;
  }
  __syncthreads();
#pragma unroll
  for (int q = 0; q < 4; ++q) {
    int n = n0 + ty + q * 8;
    int k = k0 + tx;
    if (n < NP) op[(long long)n * K + k] = f2bf(tile[tx][ty + q * 8]);
  }
}

__global__ void pad_bo_kernel(const float* __restrict__ bo, float* __restrict__ out) {
  int i = blockIdx.x * blockDim.x + threadIdx.x;
  if (i >= M_EXP * CPAD) return;
  int c = i & (CPAD - 1);
  int m = i >> 10;
  out[i] = (c < C_DIM) ? bo[m * C_DIM + c] : 0.f;
}

// ---------------- batched bf16 WMMA GEMM, bias (+ReLU) epilogue ----------------
// C[m][row,col] = A[m][row,:K] * Bt[m][col,:K]^T + bias[m][col]
// A row-major, Bt N-major (pre-transposed weights). 256 thr = 8 waves,
// 128x128 tile, BK=32, double-buffered LDS, ALL staging via async-to-LDS.
__global__ __launch_bounds__(256)
void gemm_bf16_wmma(const unsigned short* __restrict__ A, long long sAb, int lda,
                    const unsigned short* __restrict__ Bt, long long sBb, int ldbt,
                    int NPb,
                    float* __restrict__ Cf, unsigned short* __restrict__ Cb,
                    long long sCb, int ldc,
                    const float* __restrict__ bias, int sBias,
                    int Ncols, int K, int relu)
{
  __shared__ unsigned short As[2][128][LPAD];   // A tiles (rows x K)
  __shared__ unsigned short Bs[2][128][LPAD];   // B tiles (cols x K)

  const int tid  = threadIdx.x;
  const int lane = tid & 31;
  const int wave = tid >> 5;
  const int wrow = wave >> 1;              // 0..3  -> 32-row strip
  const int wcol = wave & 1;               // 0..1  -> 64-col strip
  const int mexp = blockIdx.z;
  const int rowBase = blockIdx.y * 128;
  const int colBase = blockIdx.x * 128;

  const unsigned short* Ab = A  + (long long)mexp * sAb;
  const unsigned short* Bb = Bt + (long long)mexp * sBb;

  // stage tile at k0 into buffer `buf`: both operands via async-to-LDS b128
  auto stage = [&](int k0, int buf) {
#pragma unroll
    for (int p = 0; p < 2; ++p) {
      int idx = (p * 256 + tid) * 8;
      int r = idx >> 5;
      int c = idx & 31;
      async_copy_b128((unsigned)(size_t)&As[buf][r][c],
                      Ab + (long long)(rowBase + r) * lda + (k0 + c));
      if (colBase + r < NPb)
        async_copy_b128((unsigned)(size_t)&Bs[buf][r][c],
                        Bb + (long long)(colBase + r) * ldbt + (k0 + c));
    }
  };

  v8f acc[2][4];
  v8f zv = {0.f,0.f,0.f,0.f,0.f,0.f,0.f,0.f};
#pragma unroll
  for (int i = 0; i < 2; ++i)
#pragma unroll
    for (int j = 0; j < 4; ++j) acc[i][j] = zv;

  const int r0 = lane & 15;
  const int kA = (lane >> 4) * 8;          // A frag: K chunks {kA..kA+7, kA+16..kA+23}
  const int kB = (lane >> 4) * 8;          // B frag: K chunks {kB..kB+7, kB+16..kB+23}

  const int ntiles = K >> 5;
  stage(0, 0);

  for (int t = 0; t < ntiles; ++t) {
    const int buf = t & 1;
    wait_async0();          // my async slices for `buf` have landed
    __syncthreads();        // everyone's slices landed; prior readers of buf^1 done

    if (t + 1 < ntiles) stage((t + 1) << 5, buf ^ 1);  // overlap with compute

    v16bf afrag[2], bfrag[4];
#pragma unroll
    for (int i = 0; i < 2; ++i) {
      int r = wrow * 32 + i * 16 + r0;
      v8bf lo = *(const v8bf*)&As[buf][r][kA];
      v8bf hi = *(const v8bf*)&As[buf][r][kA + 16];
#pragma unroll
      for (int j = 0; j < 8; ++j) { afrag[i][j] = lo[j]; afrag[i][j + 8] = hi[j]; }
    }
#pragma unroll
    for (int j = 0; j < 4; ++j) {
      int n = wcol * 64 + j * 16 + r0;
      v8bf lo = *(const v8bf*)&Bs[buf][n][kB];
      v8bf hi = *(const v8bf*)&Bs[buf][n][kB + 16];
#pragma unroll
      for (int q = 0; q < 8; ++q) { bfrag[j][q] = lo[q]; bfrag[j][q + 8] = hi[q]; }
    }
#pragma unroll
    for (int i = 0; i < 2; ++i)
#pragma unroll
      for (int j = 0; j < 4; ++j)
        acc[i][j] = __builtin_amdgcn_wmma_f32_16x16x32_bf16(
            false, afrag[i], false, bfrag[j], (short)0, acc[i][j], false, false);
  }

  // epilogue: C/D layout -> row = vgpr + 8*(lane>=16), col = lane&15
  const float* bp = bias + mexp * sBias;
  const int rsel = (lane >> 4) * 8;
#pragma unroll
  for (int i = 0; i < 2; ++i) {
#pragma unroll
    for (int j = 0; j < 4; ++j) {
      int colg = colBase + wcol * 64 + j * 16 + (lane & 15);
      if (colg < Ncols) {
        float bv = bp[colg];
#pragma unroll
        for (int v = 0; v < 8; ++v) {
          int rowg = rowBase + wrow * 32 + i * 16 + rsel + v;
          float val = acc[i][j][v] + bv;
          if (relu) val = fmaxf(val, 0.f);
          long long off = (long long)mexp * sCb + (long long)rowg * ldc + colg;
          if (Cf) Cf[off] = val;
          else    Cb[off] = f2bf(val);
        }
      }
    }
  }
}

// ---------------- softmax + power iteration (thread per sample) ----------------
__global__ void power_kernel(const float* __restrict__ dl,
                             float* __restrict__ pw, float* __restrict__ logw) {
  int b = blockIdx.x * blockDim.x + threadIdx.x;
  if (b >= B_DIM) return;
  const float* D = dl + (long long)b * (M_EXP * M_EXP);
  float rmax[16], rinv[16];
#pragma unroll
  for (int m = 0; m < 16; ++m) {
    float mx = D[m * 16];
#pragma unroll
    for (int n = 1; n < 16; ++n) mx = fmaxf(mx, D[m * 16 + n]);
    float s = 0.f;
#pragma unroll
    for (int n = 0; n < 16; ++n) s += __expf(D[m * 16 + n] - mx);
    rmax[m] = mx; rinv[m] = 1.f / s;
  }
  float p[16];
#pragma unroll
  for (int n = 0; n < 16; ++n) p[n] = 1.f / 16.f;
  for (int it = 0; it < PITER; ++it) {
    float pn[16];
#pragma unroll
    for (int n = 0; n < 16; ++n) pn[n] = 0.f;
#pragma unroll
    for (int m = 0; m < 16; ++m) {
      float w = p[m] * rinv[m];
#pragma unroll
      for (int n = 0; n < 16; ++n) pn[n] += w * __expf(D[m * 16 + n] - rmax[m]);
    }
    float s = 0.f;
#pragma unroll
    for (int n = 0; n < 16; ++n) s += pn[n];
    float inv = 1.f / s;
#pragma unroll
    for (int n = 0; n < 16; ++n) p[n] = pn[n] * inv;
  }
#pragma unroll
  for (int n = 0; n < 16; ++n) {
    pw[b * 16 + n]   = p[n];
    logw[b * 16 + n] = __logf(p[n] + 1e-9f);
  }
}

// ---------------- per-(b,m) logsumexp over classes: t = logw - lse ----------------
__global__ void lse_kernel(const float* __restrict__ ys, float* __restrict__ t) {
  int w = blockIdx.x * 8 + (threadIdx.x >> 5);
  int lane = threadIdx.x & 31;
  if (w >= B_DIM * M_EXP) return;
  const float* row = ys + (long long)w * CPAD;
  float mx = -3.402823466e+38f;
  for (int c = lane; c < C_DIM; c += 32) mx = fmaxf(mx, row[c]);
#pragma unroll
  for (int o = 16; o > 0; o >>= 1) mx = fmaxf(mx, __shfl_xor(mx, o, 32));
  float s = 0.f;
  for (int c = lane; c < C_DIM; c += 32) s += __expf(row[c] - mx);
#pragma unroll
  for (int o = 16; o > 0; o >>= 1) s += __shfl_xor(s, o, 32);
  if (lane == 0) t[w] -= (mx + __logf(s));
}

// ---------------- final stable mixture: y[b,c] = lse_m(ys + t) ----------------
__global__ void mix_kernel(const float* __restrict__ ys, const float* __restrict__ t,
                           float* __restrict__ y) {
  long long i = (long long)blockIdx.x * blockDim.x + threadIdx.x;
  if (i >= (long long)B_DIM * C_DIM) return;
  int b = (int)(i / C_DIM);
  int c = (int)(i - (long long)b * C_DIM);
  const float* yb = ys + (long long)b * M_EXP * CPAD + c;
  const float* tb = t + b * M_EXP;
  float mx = -3.402823466e+38f;
#pragma unroll
  for (int m = 0; m < 16; ++m) mx = fmaxf(mx, yb[m * CPAD] + tb[m]);
  float s = 0.f;
#pragma unroll
  for (int m = 0; m < 16; ++m) s += __expf(yb[m * CPAD] + tb[m] - mx);
  y[i] = mx + __logf(s);
}

extern "C" void kernel_launch(void* const* d_in, const int* in_sizes, int n_in,
                              void* d_out, int out_size, void* d_ws, size_t ws_size,
                              hipStream_t stream) {
  (void)in_sizes; (void)n_in; (void)out_size; (void)ws_size;
  const float* x  = (const float*)d_in[0];
  const float* W1 = (const float*)d_in[1];
  const float* b1 = (const float*)d_in[2];
  const float* W2 = (const float*)d_in[3];
  const float* b2 = (const float*)d_in[4];
  const float* Wo = (const float*)d_in[5];
  const float* bo = (const float*)d_in[6];
  const float* Wd = (const float*)d_in[7];
  const float* bd = (const float*)d_in[8];
  float* out = (float*)d_out;

  char* ws = (char*)d_ws;
  size_t off = 0;
  auto take = [&](size_t bytes) -> void* {
    void* p = ws + off;
    off += (bytes + 255) & ~(size_t)255;
    return p;
  };
  unsigned short* xb   = (unsigned short*)take((size_t)B_DIM * DIN * 2);
  unsigned short* W1t  = (unsigned short*)take((size_t)M_EXP * H_DIM * DIN * 2);  // [m][2048][1024]
  unsigned short* W2t  = (unsigned short*)take((size_t)M_EXP * H_DIM * H_DIM * 2);// [m][2048][2048]
  unsigned short* Wot  = (unsigned short*)take((size_t)M_EXP * CPAD * H_DIM * 2); // [m][1024][2048]
  unsigned short* Wdt  = (unsigned short*)take((size_t)M_EXP * M_EXP * H_DIM * 2);// [m][16][2048]
  float*          bop  = (float*)take((size_t)M_EXP * CPAD * 4);
  unsigned short* h1   = (unsigned short*)take((size_t)B_DIM * M_EXP * H_DIM * 2);
  unsigned short* h2   = (unsigned short*)take((size_t)B_DIM * M_EXP * H_DIM * 2);
  float*          ys   = (float*)h1;  // alias: h1 dead after stage 2, same byte size
  float*          dl   = (float*)take((size_t)B_DIM * M_EXP * M_EXP * 4);
  float*          logw = (float*)take((size_t)B_DIM * M_EXP * 4);

  const int T = 256;
  long long nx = (long long)B_DIM * DIN / 8;
  cvt_bf16x8_kernel<<<(int)((nx + T - 1) / T), T, 0, stream>>>(x, xb, nx);

  // weights: transpose + convert + pad (one-shot)
  tr_cvt_kernel<<<dim3(H_DIM / 32, DIN / 32, M_EXP), 256, 0, stream>>>(
      W1, (long long)DIN * H_DIM, W1t, (long long)H_DIM * DIN, DIN, H_DIM, H_DIM);
  tr_cvt_kernel<<<dim3(H_DIM / 32, H_DIM / 32, M_EXP), 256, 0, stream>>>(
      W2, (long long)H_DIM * H_DIM, W2t, (long long)H_DIM * H_DIM, H_DIM, H_DIM, H_DIM);
  tr_cvt_kernel<<<dim3(CPAD / 32, H_DIM / 32, M_EXP), 256, 0, stream>>>(
      Wo, (long long)H_DIM * C_DIM, Wot, (long long)CPAD * H_DIM, H_DIM, C_DIM, CPAD);
  tr_cvt_kernel<<<dim3(1, H_DIM / 32, M_EXP), 256, 0, stream>>>(
      Wd, (long long)H_DIM * M_EXP, Wdt, (long long)M_EXP * H_DIM, H_DIM, M_EXP, M_EXP);
  pad_bo_kernel<<<(M_EXP * CPAD + T - 1) / T, T, 0, stream>>>(bo, bop);

  // stage 1: h1 = relu(x @ W1 + b1)
  gemm_bf16_wmma<<<dim3(H_DIM / 128, B_DIM / 128, M_EXP), 256, 0, stream>>>(
      xb, 0LL, DIN,
      W1t, (long long)H_DIM * DIN, DIN, H_DIM,
      nullptr, h1, (long long)H_DIM, M_EXP * H_DIM,
      b1, H_DIM, H_DIM, DIN, 1);
  // stage 2: h2 = relu(h1 @ W2 + b2)
  gemm_bf16_wmma<<<dim3(H_DIM / 128, B_DIM / 128, M_EXP), 256, 0, stream>>>(
      h1, (long long)H_DIM, M_EXP * H_DIM,
      W2t, (long long)H_DIM * H_DIM, H_DIM, H_DIM,
      nullptr, h2, (long long)H_DIM, M_EXP * H_DIM,
      b2, H_DIM, H_DIM, H_DIM, 1);
  // stage 3: ys = h2 @ Wo + bo (fp32 out)
  gemm_bf16_wmma<<<dim3(CPAD / 128, B_DIM / 128, M_EXP), 256, 0, stream>>>(
      h2, (long long)H_DIM, M_EXP * H_DIM,
      Wot, (long long)CPAD * H_DIM, H_DIM, CPAD,
      ys, nullptr, (long long)CPAD, M_EXP * CPAD,
      bop, CPAD, CPAD, H_DIM, 0);
  // stage 4: dl = h2 @ Wd + bd (fp32 out)
  gemm_bf16_wmma<<<dim3(1, B_DIM / 128, M_EXP), 256, 0, stream>>>(
      h2, (long long)H_DIM, M_EXP * H_DIM,
      Wdt, (long long)M_EXP * H_DIM, H_DIM, M_EXP,
      dl, nullptr, (long long)M_EXP, M_EXP * M_EXP,
      bd, M_EXP, M_EXP, H_DIM, 0);

  float* power_out = out + (long long)B_DIM * C_DIM;   // tuple: (y, power)
  power_kernel<<<(B_DIM + T - 1) / T, T, 0, stream>>>(dl, power_out, logw);
  lse_kernel<<<(B_DIM * M_EXP) / 8, 256, 0, stream>>>(ys, logw);
  long long ny = (long long)B_DIM * C_DIM;
  mix_kernel<<<(int)((ny + T - 1) / T), T, 0, stream>>>(ys, logw, out);
}
